// Vig_29008209117773
// MI455X (gfx1250) — matrix operand order repaired and checked by reading, not compiled, not run on hardware
//
#include <hip/hip_runtime.h>

#define B_   32
#define N_   1024
#define FIN  512
#define FOUT 256
#define KNN  16
#define DSTR 1028   // padded LDS row stride (floats) for the distance block

typedef __attribute__((ext_vector_type(16))) _Float16 v16h;
typedef __attribute__((ext_vector_type(8)))  float    v8f;
typedef __attribute__((address_space(3))) unsigned char* lds_raw_t;

__device__ __forceinline__ float wave_sum(float v) {
  #pragma unroll
  for (int off = 16; off > 0; off >>= 1) v += __shfl_xor(v, off, 32);
  return v;
}
__device__ __forceinline__ float wave_max(float v) {
  #pragma unroll
  for (int off = 16; off > 0; off >>= 1) v = fmaxf(v, __shfl_xor(v, off, 32));
  return v;
}

// B fragment (32x16 f16, column-per-lane): 8 halfs at p[k0+kb], 8 at p[k0+16+kb].
__device__ __forceinline__ v16h load_bfrag(const _Float16* __restrict__ p, int k0kb) {
  v16h b;
  #pragma unroll
  for (int e = 0; e < 16; ++e) b[e] = p[k0kb + e + (e >= 8 ? 8 : 0)];
  return b;
}
// A fragment (16x32 f16, row-in-lane) from an LDS tile of shape [16][FIN].
__device__ __forceinline__ v16h load_afrag(const _Float16* __restrict__ At, int row,
                                           int k0, int half8) {
  v16h a;
  #pragma unroll
  for (int e = 0; e < 8; ++e) {
    a[e]     = At[row * FIN + k0 + half8 + e];
    a[e + 8] = At[row * FIN + k0 + 16 + half8 + e];
  }
  return a;
}
__device__ __forceinline__ v8f wmma_f16(v16h a, v16h b, v8f c) {
  return __builtin_amdgcn_wmma_f32_16x16x32_f16(false, a, false, b, (short)0, c,
                                                false, false);
}

// Stage `bytes16` 16-byte chunks from global -> LDS via the gfx1250 async path.
// Ends with s_wait_asynccnt 0; all callers follow with __syncthreads().
__device__ __forceinline__ void async_tile_to_lds(_Float16* dstLds,
                                                  const _Float16* __restrict__ src,
                                                  int nchunks, int tid, int nthr) {
  unsigned int ldsbase = (unsigned int)(unsigned long long)(lds_raw_t)(void*)dstLds;
  unsigned long long gbase = (unsigned long long)(const void*)src;
  for (int i = tid; i < nchunks; i += nthr) {
    unsigned int       laddr = ldsbase + (unsigned int)(i * 16);
    unsigned long long gaddr = gbase + (unsigned long long)(i * 16);
    asm volatile("global_load_async_to_lds_b128 %0, %1, off"
                 :: "v"(laddr), "v"(gaddr) : "memory");
  }
  asm volatile("s_wait_asynccnt 0x0" ::: "memory");
}

// ---------------------------------------------------------------------------
// Kernel 1: x (f32) -> f16 copy + per-node squared norm. One wave per node.
// ---------------------------------------------------------------------------
__global__ __launch_bounds__(256)
void cvt_x_kernel(const float* __restrict__ x, _Float16* __restrict__ xh,
                  float* __restrict__ x2g) {
  const int wave = (blockIdx.x * blockDim.x + threadIdx.x) >> 5;
  const int lane = threadIdx.x & 31;
  if (wave >= B_ * N_) return;
  const float* row = x + (size_t)wave * FIN;
  _Float16*   orow = xh + (size_t)wave * FIN;
  float s = 0.f;
  #pragma unroll 4
  for (int k = lane; k < FIN; k += 32) {
    float v = row[k];
    s += v * v;
    orow[k] = (_Float16)v;
  }
  s = wave_sum(s);
  if (lane == 0) x2g[wave] = s;
}

// ---------------------------------------------------------------------------
// Kernel 1b: W (f32, [FIN][FOUT]) -> wt (f16, transposed [FOUT][FIN]).
// One wave per output column. Makes GEMM B fragments contiguous b128 loads.
// ---------------------------------------------------------------------------
__global__ __launch_bounds__(256)
void cvt_w_kernel(const float* __restrict__ W, _Float16* __restrict__ wt) {
  const int c    = (blockIdx.x * blockDim.x + threadIdx.x) >> 5;
  const int lane = threadIdx.x & 31;
  if (c >= FOUT) return;
  #pragma unroll 4
  for (int k = lane; k < FIN; k += 32)
    wt[(size_t)c * FIN + k] = (_Float16)W[(size_t)k * FOUT + c];
}

// ---------------------------------------------------------------------------
// Kernel 2: kNN. One WG (16 waves) per (batch, 16-row block).
// Phase 1: WMMA Gram tiles (dual accumulators, pipelined B loads)
//          -> 16 x 1024 distance block in padded dynamic LDS.
// Phase 2: wave w does 16 rounds of wave-parallel argmin for row w.
// Dyn LDS: A-tile 16KB + x2row 64B + dist 16*1028*4 = 82240 B (< 320KB WGP).
// ---------------------------------------------------------------------------
extern __shared__ __align__(16) unsigned char smem_dyn[];

__global__ __launch_bounds__(512)
void knn_kernel(const _Float16* __restrict__ xh, const float* __restrict__ x2g,
                int* __restrict__ knn) {
  const int b    = blockIdx.x >> 6;     // / (N_/16)
  const int rb   = blockIdx.x & 63;
  const int tid  = threadIdx.x;
  const int lane = tid & 31;
  const int wave = tid >> 5;

  _Float16* At    = (_Float16*)smem_dyn;                       // 16 x 512 f16
  float*    x2row = (float*)(smem_dyn + 16 * FIN * 2);         // 16 f32
  float*    dist  = (float*)(smem_dyn + 16 * FIN * 2 + 64);    // 16 x DSTR f32

  const _Float16* arow = xh + ((size_t)b * N_ + (size_t)rb * 16) * FIN;
  async_tile_to_lds(At, arow, (16 * FIN * 2) / 16, tid, 512);
  if (tid < 16) x2row[tid] = x2g[b * N_ + rb * 16 + tid];
  __syncthreads();

  const int row   = lane & 15;            // A row this lane feeds
  const int half8 = (lane & 16) ? 8 : 0;  // A K-half offset; also C/D M offset
  const int kb    = (lane & 16) ? 8 : 0;  // B K-half offset

  // Phase 1: each wave computes 2 pairs of adjacent 16x16 tiles.
  #pragma unroll
  for (int pr = 0; pr < 2; ++pr) {
    const int cb0 = wave * 2 + pr * 32;          // even block in [0,62]
    const int c0  = cb0 * 16 + (lane & 15);
    const int c1  = c0 + 16;
    const _Float16* brow0 = xh + ((size_t)b * N_ + c0) * FIN;
    const _Float16* brow1 = xh + ((size_t)b * N_ + c1) * FIN;

    v16h b0n = load_bfrag(brow0, kb);
    v16h b1n = load_bfrag(brow1, kb);
    v8f acc0 = {}, acc1 = {};
    for (int k0 = 0; k0 < FIN; k0 += 32) {
      const int kn = (k0 + 32) & (FIN - 1);      // wraps; extra load is harmless
      v16h b0 = b0n, b1 = b1n;
      b0n = load_bfrag(brow0, kn + kb);          // prefetch next k-step
      b1n = load_bfrag(brow1, kn + kb);
      v16h a = load_afrag(At, row, k0, half8);
      acc0 = wmma_f16(a, b0, acc0);
      acc1 = wmma_f16(a, b1, acc1);
    }
    const float xc0 = x2g[b * N_ + c0];
    const float xc1 = x2g[b * N_ + c1];
    #pragma unroll
    for (int r = 0; r < 8; ++r) {
      const int m = r + half8;
      dist[m * DSTR + c0] = x2row[m] + xc0 - 2.0f * acc0[r];
      dist[m * DSTR + c1] = x2row[m] + xc1 - 2.0f * acc1[r];
    }
  }
  __syncthreads();

  // Phase 2: top-16 (smallest distance) for row `wave` (16 waves, 16 rows)
  float* drow = dist + wave * DSTR;
  int*   orow = knn + ((size_t)b * N_ + (size_t)rb * 16 + wave) * KNN;
  for (int p = 0; p < KNN; ++p) {
    float bv = 3.0e38f;
    int   bi = N_;
    for (int j = lane; j < N_; j += 32) {
      float v = drow[j];
      if (v < bv) { bv = v; bi = j; }
    }
    #pragma unroll
    for (int off = 16; off > 0; off >>= 1) {
      float ov = __shfl_xor(bv, off, 32);
      int   oi = __shfl_xor(bi, off, 32);
      if (ov < bv || (ov == bv && oi < bi)) { bv = ov; bi = oi; }
    }
    if (lane == 0) { orow[p] = bi; drow[bi] = 3.0e38f; }
    __syncthreads();   // make lane0's LDS kill visible before the next pass
  }
}

// ---------------------------------------------------------------------------
// Kernel 3: h = x @ W via WMMA. One WG (8 waves) per 16-row block;
// each wave owns 2 adjacent output column tiles (dual acc, pipelined loads).
// ---------------------------------------------------------------------------
__global__ __launch_bounds__(256)
void gemm_h_kernel(const _Float16* __restrict__ xh, const _Float16* __restrict__ wt,
                   float* __restrict__ h) {
  const int b    = blockIdx.x >> 6;
  const int rb   = blockIdx.x & 63;
  const int tid  = threadIdx.x;
  const int lane = tid & 31;
  const int wave = tid >> 5;

  __shared__ _Float16 At[16 * FIN];
  const _Float16* arow = xh + ((size_t)b * N_ + (size_t)rb * 16) * FIN;
  async_tile_to_lds(At, arow, (16 * FIN * 2) / 16, tid, 256);
  __syncthreads();

  const int row   = lane & 15;
  const int half8 = (lane & 16) ? 8 : 0;
  const int kb    = (lane & 16) ? 8 : 0;

  const int c0 = wave * 32 + (lane & 15);   // wave owns col tiles 2w, 2w+1
  const int c1 = c0 + 16;
  const _Float16* brow0 = wt + (size_t)c0 * FIN;
  const _Float16* brow1 = wt + (size_t)c1 * FIN;

  v16h b0n = load_bfrag(brow0, kb);
  v16h b1n = load_bfrag(brow1, kb);
  v8f acc0 = {}, acc1 = {};
  for (int k0 = 0; k0 < FIN; k0 += 32) {
    const int kn = (k0 + 32) & (FIN - 1);
    v16h b0 = b0n, b1 = b1n;
    b0n = load_bfrag(brow0, kn + kb);
    b1n = load_bfrag(brow1, kn + kb);
    v16h a = load_afrag(At, row, k0, half8);
    acc0 = wmma_f16(a, b0, acc0);
    acc1 = wmma_f16(a, b1, acc1);
  }
  #pragma unroll
  for (int r = 0; r < 8; ++r) {
    const size_t orow = ((size_t)b * N_ + (size_t)rb * 16 + r + half8) * FOUT;
    h[orow + c0] = acc0[r];
    h[orow + c1] = acc1[r];
  }
}

// ---------------------------------------------------------------------------
// Kernel 4: f_src / f_dst attention projections. One wave per node.
// ---------------------------------------------------------------------------
__global__ __launch_bounds__(256)
void f_kernel(const float* __restrict__ h, const float* __restrict__ a_src,
              const float* __restrict__ a_dst, float* __restrict__ fsrc,
              float* __restrict__ fdst) {
  const int wave = (blockIdx.x * blockDim.x + threadIdx.x) >> 5;
  const int lane = threadIdx.x & 31;
  if (wave >= B_ * N_) return;
  const float* hrow = h + (size_t)wave * FOUT;
  float s1 = 0.f, s2 = 0.f;
  #pragma unroll
  for (int e = lane; e < FOUT; e += 32) {
    float v = hrow[e];
    s1 += v * a_src[e];
    s2 += v * a_dst[e];
  }
  s1 = wave_sum(s1);
  s2 = wave_sum(s2);
  if (lane == 0) { fsrc[wave] = s1; fdst[wave] = s2; }
}

// ---------------------------------------------------------------------------
// Kernel 5: sparse softmax over the 16 neighbors + aggregation + ELU.
// One wave per node; lane L owns output features [8L, 8L+8).
// (Dense softmax with -1e9 mask == softmax over the K neighbors: exp of the
//  masked entries underflows to exactly 0 in f32.)
// ---------------------------------------------------------------------------
__global__ __launch_bounds__(256)
void agg_kernel(const float* __restrict__ h, const int* __restrict__ knn,
                const float* __restrict__ fsrc, const float* __restrict__ fdst,
                float* __restrict__ out) {
  const int node = (blockIdx.x * blockDim.x + threadIdx.x) >> 5;
  const int lane = threadIdx.x & 31;
  if (node >= B_ * N_) return;
  const int b = node >> 10;  // / N_

  const int* nb = knn + (size_t)node * KNN;
  const float fs = fsrc[node];

  float ej = -3.0e38f;
  int   nj = 0;
  if (lane < KNN) {
    nj = nb[lane];
    float v = fs + fdst[b * N_ + nj];
    ej = (v > 0.f) ? v : 0.2f * v;   // LeakyReLU(0.2)
  }
  const float m = wave_max(ej);
  float w = (lane < KNN) ? __expf(ej - m) : 0.f;
  const float s = wave_sum(w);
  const float attn = w / s;

  float acc[8];
  #pragma unroll
  for (int t = 0; t < 8; ++t) acc[t] = 0.f;

  for (int j = 0; j < KNN; ++j) {
    const int   idxj = __shfl(nj, j, 32);
    const float aj   = __shfl(attn, j, 32);
    const float* hr = h + ((size_t)b * N_ + idxj) * FOUT + lane * 8;
    #pragma unroll
    for (int t = 0; t < 8; ++t) acc[t] += aj * hr[t];
  }

  float* orow = out + (size_t)node * FOUT + lane * 8;
  #pragma unroll
  for (int t = 0; t < 8; ++t) {
    float v = acc[t];
    orow[t] = (v > 0.f) ? v : (__expf(v) - 1.0f);  // ELU
  }
}

// ---------------------------------------------------------------------------
extern "C" void kernel_launch(void* const* d_in, const int* in_sizes, int n_in,
                              void* d_out, int out_size, void* d_ws, size_t ws_size,
                              hipStream_t stream) {
  const float* x     = (const float*)d_in[0];  // [B, N, FIN]
  const float* W     = (const float*)d_in[1];  // [FIN, FOUT]
  const float* a_src = (const float*)d_in[2];  // [FOUT]
  const float* a_dst = (const float*)d_in[3];  // [FOUT]
  float*       out   = (float*)d_out;          // [B, N, FOUT]

  // Workspace layout
  unsigned char* ws = (unsigned char*)d_ws;
  size_t off = 0;
  _Float16* xh = (_Float16*)(ws + off); off += (size_t)B_ * N_ * FIN * sizeof(_Float16);
  _Float16* wt = (_Float16*)(ws + off); off += (size_t)FOUT * FIN * sizeof(_Float16);
  float* x2g   = (float*)(ws + off);    off += (size_t)B_ * N_ * sizeof(float);
  float* h     = (float*)(ws + off);    off += (size_t)B_ * N_ * FOUT * sizeof(float);
  float* fsrc  = (float*)(ws + off);    off += (size_t)B_ * N_ * sizeof(float);
  float* fdst  = (float*)(ws + off);    off += (size_t)B_ * N_ * sizeof(float);
  int*   knn   = (int*)(ws + off);      off += (size_t)B_ * N_ * KNN * sizeof(int);
  (void)off; (void)ws_size; (void)in_sizes; (void)n_in; (void)out_size;

  const int nodes = B_ * N_;

  // 1) conversions
  cvt_x_kernel<<<nodes / 8, 256, 0, stream>>>(x, xh, x2g);
  cvt_w_kernel<<<FOUT / 8, 256, 0, stream>>>(W, wt);

  // 2) kNN distances (WMMA) + top-16 selection
  const size_t knn_lds = (size_t)16 * FIN * 2 + 64 + (size_t)16 * DSTR * 4; // 82240 B
  knn_kernel<<<B_ * (N_ / 16), 512, knn_lds, stream>>>(xh, x2g, knn);

  // 3) h = x @ W (WMMA)
  gemm_h_kernel<<<B_ * (N_ / 16), 256, 0, stream>>>(xh, wt, h);

  // 4) attention projections
  f_kernel<<<nodes / 8, 256, 0, stream>>>(h, a_src, a_dst, fsrc, fdst);

  // 5) sparse softmax + aggregation + ELU
  agg_kernel<<<nodes / 8, 256, 0, stream>>>(h, knn, fsrc, fdst, out);
}
